// MambaEncoderLayer_40097814675763
// MI455X (gfx1250) — matrix operand order repaired and checked by reading, not compiled, not run on hardware
//
#include <hip/hip_runtime.h>
#include <math.h>

// ---------------------------------------------------------------------------
// Mamba encoder layer for MI455X (gfx1250, wave32, WMMA + async LDS DMA).
//   fp32->bf16 once per operand, then all projections run as bf16 WMMA GEMMs
//   (fp32 accumulate) fed by GLOBAL_LOAD_ASYNC_TO_LDS_B128 double buffering.
//   Scan = 1 state/lane, 16 lanes/channel, shfl_xor reduce, fused skip+gate.
// ---------------------------------------------------------------------------

typedef __attribute__((ext_vector_type(16))) __bf16 v16bf;
typedef __attribute__((ext_vector_type(8)))  float  v8f;
typedef __attribute__((ext_vector_type(4)))  __bf16 bf16x4;
typedef int v4i_gcc __attribute__((vector_size(16)));   // matches builtin param

#define LDT 40   // LDS tile row stride in bf16 (80B: 16B-aligned, bank-friendly)

#if __has_builtin(__builtin_amdgcn_global_load_async_to_lds_b128)
#define HAVE_ASYNC_LDS 1
#else
#define HAVE_ASYNC_LDS 0
#endif

__device__ __forceinline__ void cp16_g2l(const __bf16* g, __bf16* l) {
#if HAVE_ASYNC_LDS
    __builtin_amdgcn_global_load_async_to_lds_b128(
        (__attribute__((address_space(1))) v4i_gcc*)g,
        (__attribute__((address_space(3))) v4i_gcc*)l, 0, 0);
#else
    *(int4*)l = *(const int4*)g;
#endif
}

__device__ __forceinline__ void wait_async_lds() {
#if HAVE_ASYNC_LDS
#if __has_builtin(__builtin_amdgcn_s_wait_asynccnt)
    __builtin_amdgcn_s_wait_asynccnt(0);
#else
    asm volatile("s_wait_asynccnt 0" ::: "memory");
#endif
#endif
}

__device__ __forceinline__ float softplus_f(float v) {
    return v > 20.0f ? v : log1pf(__expf(v));
}
__device__ __forceinline__ float silu_f(float v) {
    return v / (1.0f + __expf(-v));
}

// fp32 -> bf16 bulk convert (n4 = count/4)
__global__ void __launch_bounds__(256)
cvt_bf16_kernel(const float* __restrict__ s, __bf16* __restrict__ d, int n4)
{
    const int i = blockIdx.x * 256 + threadIdx.x;
    if (i < n4) {
        float4 v = ((const float4*)s)[i];
        bf16x4 o;
        o[0] = (__bf16)v.x; o[1] = (__bf16)v.y;
        o[2] = (__bf16)v.z; o[3] = (__bf16)v.w;
        ((bf16x4*)d)[i] = o;
    }
}

// C[m,n] = sum_k A[m,k]*W[n,k] (+bias) (opt softplus), bf16 inputs, fp32 acc.
// Requires: K % 32 == 0, M % 64 == 0 (true at all call sites).
template<bool SOFTPLUS, bool RAGGED_N>
__global__ void __launch_bounds__(128)
gemm_bf16_wmma(const __bf16* __restrict__ A, int lda,
               const __bf16* __restrict__ W, int ldw,
               const float* __restrict__ bias,
               float* __restrict__ C, __bf16* __restrict__ Cb, int ldc,
               int N, int K)
{
    __shared__ __bf16 sA[2][64][LDT];
    __shared__ __bf16 sB[2][64][LDT];

    const int tid  = threadIdx.x;
    const int lane = tid & 31;
    const int wv   = tid >> 5;
    const int wm   = wv >> 1;       // wave row-block (32 rows)
    const int wn   = wv & 1;        // wave col-block (32 cols)
    const int hl   = lane >> 4;
    const int l16  = lane & 15;

    const int mbase = blockIdx.y * 64;
    const int nbase = blockIdx.x * 64;

    v8f acc[2][2] = {};

    // async loader: 64x32 bf16 tile = 256 x 16B chunks; 128 thr x 2 chunks
    const int row0 = tid >> 2;      // 0..31 (+32 for 2nd chunk)
    const int sub  = tid & 3;       // 16B chunk within a 64B row

    auto issue = [&](int b, int k0) {
        #pragma unroll
        for (int cc = 0; cc < 2; ++cc) {
            const int row = row0 + cc * 32;
            cp16_g2l(A + (size_t)(mbase + row) * lda + k0 + sub * 8,
                     &sA[b][row][sub * 8]);
            const int n = nbase + row;
            if (!RAGGED_N || n < N)
                cp16_g2l(W + (size_t)n * ldw + k0 + sub * 8,
                         &sB[b][row][sub * 8]);
            else
                *(int4*)&sB[b][row][sub * 8] = make_int4(0, 0, 0, 0);
        }
    };

    issue(0, 0);
    int buf = 0;
    for (int k0 = 0; k0 < K; k0 += 32, buf ^= 1) {
        wait_async_lds();
        __syncthreads();
        if (k0 + 32 < K) issue(buf ^ 1, k0 + 32);   // DMA overlaps WMMA below

        // CDNA5 fragment layouts (16-bit 16x32 A / 32x16 B)
        v16bf afrag[2], bfrag[2];
        #pragma unroll
        for (int s = 0; s < 2; ++s) {
            const int arow = wm * 32 + s * 16 + l16;
            #pragma unroll
            for (int j = 0; j < 8; ++j) {
                // lanes 0-15: V0..3->K0..7, V4..7->K16..23; lanes 16-31: +8
                int k = (j < 4 ? 2 * j : 16 + 2 * (j - 4)) + 8 * hl;
                afrag[s][2 * j]     = sA[buf][arow][k];
                afrag[s][2 * j + 1] = sA[buf][arow][k + 1];
            }
            const int brow = wn * 32 + s * 16 + l16;    // output column n
            #pragma unroll
            for (int i = 0; i < 16; ++i)                // lanes 0-15: K0..7,16..23; lanes 16-31: +8
                bfrag[s][i] = sB[buf][brow][(i < 8 ? i : 8 + i) + 8 * hl];
        }
        #pragma unroll
        for (int i = 0; i < 2; ++i)
            #pragma unroll
            for (int j = 0; j < 2; ++j)
                acc[i][j] = __builtin_amdgcn_wmma_f32_16x16x32_bf16(
                    false, afrag[i], false, bfrag[j],
                    (short)0, acc[i][j], false, false);
    }

    // epilogue: C/D layout -> M = r + 8*hl, N = lane&15
    #pragma unroll
    for (int i = 0; i < 2; ++i) {
        #pragma unroll
        for (int j = 0; j < 2; ++j) {
            const int n = nbase + wn * 32 + j * 16 + l16;
            if (RAGGED_N && n >= N) continue;
            const float bb = (bias != nullptr) ? bias[n] : 0.0f;
            #pragma unroll
            for (int r = 0; r < 8; ++r) {
                const int m = mbase + wm * 32 + i * 16 + r + 8 * hl;
                float v = acc[i][j][r] + bb;
                if (SOFTPLUS) v = softplus_f(v);
                C[(size_t)m * ldc + n] = v;
                if (Cb != nullptr) Cb[(size_t)m * ldc + n] = (__bf16)v;
            }
        }
    }
}

// Depthwise causal conv (k=4) + bias + SiLU; emits fp32 (scan) + bf16 (GEMM).
__global__ void __launch_bounds__(256)
conv_silu_kernel(const float* __restrict__ xz,
                 const float* __restrict__ cw,
                 const float* __restrict__ cb,
                 float* __restrict__ xi, __bf16* __restrict__ xib)
{
    const int idx = blockIdx.x * 256 + threadIdx.x;   // over 4096*2048
    const int d   = idx & 2047;
    const int row = idx >> 11;                        // b*2048 + t
    const int t   = row & 2047;
    float acc = cb[d];
    #pragma unroll
    for (int j = 0; j < 4; ++j) {
        int tt = t - 3 + j;
        if (tt >= 0)
            acc += cw[d * 4 + j] * xz[(size_t)(row - 3 + j) * 4096 + d];
    }
    const float s = silu_f(acc);
    xi[(size_t)row * 2048 + d]  = s;
    xib[(size_t)row * 2048 + d] = (__bf16)s;
}

// Selective scan: one state n per lane, 16 lanes per (b,d) channel.
// Fuses D*x skip and SiLU(z) gate; writes bf16 (feeds out_proj directly).
__global__ void __launch_bounds__(256)
scan_kernel(const float* __restrict__ dtf,  // [4096,2048]
            const float* __restrict__ xdbl, // [4096,96]  (dt|B|C)
            const float* __restrict__ xi,   // [4096,2048]
            const float* __restrict__ xz,   // [4096,4096] (z = cols 2048+)
            const float* __restrict__ A_log,
            const float* __restrict__ Dp,
            __bf16* __restrict__ yg)        // [4096,2048]
{
    const int n   = threadIdx.x & 15;
    const int cib = threadIdx.x >> 4;
    const int c   = blockIdx.x * 16 + cib;       // 0..4095
    const int b   = c >> 11;
    const int d   = c & 2047;

    const float a  = -__expf(A_log[d * 16 + n]); // A = -exp(A_log)
    const float Dd = Dp[d];
    const size_t rbase = (size_t)b * 2048;

    const float* pdt = dtf  + rbase * 2048 + d;
    const float* pxi = xi   + rbase * 2048 + d;
    const float* pB  = xdbl + rbase * 96 + 64 + n;
    const float* pz  = xz   + rbase * 4096 + 2048 + d;
    __bf16*      py  = yg   + rbase * 2048 + d;

    float h = 0.0f;
    for (int t = 0; t < 2048; ++t) {
        if (t + 16 < 2048) {                 // hide latency of the serial walk
            __builtin_prefetch(pdt + 16 * 2048, 0, 1);
            __builtin_prefetch(pxi + 16 * 2048, 0, 1);
        }
        const float dt = *pdt;
        const float xv = *pxi;
        const float Bn = pB[0];
        const float Cn = pB[16];
        h = h * __expf(dt * a) + (dt * xv) * Bn;
        float p = h * Cn;
        p += __shfl_xor(p, 1);
        p += __shfl_xor(p, 2);
        p += __shfl_xor(p, 4);
        p += __shfl_xor(p, 8);
        if (n == 0) {
            const float zv = *pz;
            *py = (__bf16)((p + Dd * xv) * silu_f(zv));
        }
        pdt += 2048; pxi += 2048; pB += 96; pz += 4096; py += 2048;
    }
}

// out = x + LayerNorm(m)*w + b, one 256-thread block per row of 1024.
__global__ void __launch_bounds__(256)
ln_residual_kernel(const float* __restrict__ x, const float* __restrict__ m,
                   const float* __restrict__ w, const float* __restrict__ bvec,
                   float* __restrict__ out)
{
    __shared__ float red[256];
    const int row = blockIdx.x;
    const int tid = threadIdx.x;
    const float* mr = m + (size_t)row * 1024;

    float v[4];
    float s = 0.0f;
    #pragma unroll
    for (int j = 0; j < 4; ++j) { v[j] = mr[tid + 256 * j]; s += v[j]; }
    red[tid] = s; __syncthreads();
    for (int off = 128; off > 0; off >>= 1) {
        if (tid < off) red[tid] += red[tid + off];
        __syncthreads();
    }
    const float mu = red[0] * (1.0f / 1024.0f);
    __syncthreads();

    float q = 0.0f;
    #pragma unroll
    for (int j = 0; j < 4; ++j) { float dd = v[j] - mu; q += dd * dd; }
    red[tid] = q; __syncthreads();
    for (int off = 128; off > 0; off >>= 1) {
        if (tid < off) red[tid] += red[tid + off];
        __syncthreads();
    }
    const float rs = rsqrtf(red[0] * (1.0f / 1024.0f) + 1e-6f);

    const float* xr = x + (size_t)row * 1024;
    float* orow = out + (size_t)row * 1024;
    #pragma unroll
    for (int j = 0; j < 4; ++j) {
        int i = tid + 256 * j;
        orow[i] = xr[i] + (v[j] - mu) * rs * w[i] + bvec[i];
    }
}

extern "C" void kernel_launch(void* const* d_in, const int* in_sizes, int n_in,
                              void* d_out, int out_size, void* d_ws, size_t ws_size,
                              hipStream_t stream)
{
    (void)in_sizes; (void)n_in; (void)out_size; (void)ws_size;
    const float* x     = (const float*)d_in[0];   // [2,2048,1024]
    const float* w_in  = (const float*)d_in[1];   // [4096,1024]
    const float* cw    = (const float*)d_in[2];   // [2048,4]
    const float* cb    = (const float*)d_in[3];   // [2048]
    const float* w_xp  = (const float*)d_in[4];   // [96,2048]
    const float* w_dt  = (const float*)d_in[5];   // [2048,64]
    const float* b_dt  = (const float*)d_in[6];   // [2048]
    const float* A_log = (const float*)d_in[7];   // [2048,16]
    const float* Dp    = (const float*)d_in[8];   // [2048]
    const float* w_out = (const float*)d_in[9];   // [1024,2048]
    const float* lnw   = (const float*)d_in[10];  // [1024]
    const float* lnb   = (const float*)d_in[11];  // [1024]
    float* out = (float*)d_out;

    // ---- workspace layout ----
    float* ws   = (float*)d_ws;
    float* xz   = ws;                             // 4096*4096 f32
    float* xi   = xz   + (size_t)4096 * 4096;     // 4096*2048 f32
    float* xdbl = xi   + (size_t)4096 * 2048;     // 4096*96   f32
    float* dtf  = xdbl + (size_t)4096 * 96;       // 4096*2048 f32
    float* mbuf = dtf  + (size_t)4096 * 2048;     // 4096*1024 f32
    __bf16* bfr = (__bf16*)(mbuf + (size_t)4096 * 1024);
    __bf16* x_bf    = bfr;                               // 4096*1024
    __bf16* win_bf  = x_bf    + (size_t)4096 * 1024;     // 4096*1024
    __bf16* wxp_bf  = win_bf  + (size_t)4096 * 1024;     // 96*2048
    __bf16* wdt_bf  = wxp_bf  + (size_t)96 * 2048;       // 2048*64
    __bf16* wout_bf = wdt_bf  + (size_t)2048 * 64;       // 1024*2048
    __bf16* xi_bf   = wout_bf + (size_t)1024 * 2048;     // 4096*2048
    __bf16* xdbl_bf = xi_bf   + (size_t)4096 * 2048;     // 4096*96
    __bf16* yg_bf   = xdbl_bf + (size_t)4096 * 96;       // 4096*2048

    // ---- 0) one-time fp32 -> bf16 conversions ----
    cvt_bf16_kernel<<<4096, 256, 0, stream>>>(x,     x_bf,    1048576);
    cvt_bf16_kernel<<<4096, 256, 0, stream>>>(w_in,  win_bf,  1048576);
    cvt_bf16_kernel<<<192,  256, 0, stream>>>(w_xp,  wxp_bf,  49152);
    cvt_bf16_kernel<<<128,  256, 0, stream>>>(w_dt,  wdt_bf,  32768);
    cvt_bf16_kernel<<<2048, 256, 0, stream>>>(w_out, wout_bf, 524288);

    const dim3 blk(128);
    // 1) in_proj: [4096,1024] x [4096,1024]^T -> xz[4096,4096]
    gemm_bf16_wmma<false, false><<<dim3(64, 64), blk, 0, stream>>>(
        x_bf, 1024, win_bf, 1024, nullptr, xz, nullptr, 4096, 4096, 1024);
    // 2) depthwise conv + SiLU (f32 + bf16 outputs)
    conv_silu_kernel<<<(4096 * 2048) / 256, 256, 0, stream>>>(xz, cw, cb, xi, xi_bf);
    // 3) x_proj: [4096,2048] x [96,2048]^T -> xdbl[4096,96] (+bf16 copy)
    gemm_bf16_wmma<false, true><<<dim3(2, 64), blk, 0, stream>>>(
        xi_bf, 2048, wxp_bf, 2048, nullptr, xdbl, xdbl_bf, 96, 96, 2048);
    // 4) dt_proj + bias + softplus: [4096,64] x [2048,64]^T -> dtf[4096,2048]
    gemm_bf16_wmma<true, false><<<dim3(32, 64), blk, 0, stream>>>(
        xdbl_bf, 96, wdt_bf, 64, b_dt, dtf, nullptr, 2048, 2048, 64);
    // 5) selective scan + skip + gate -> yg_bf
    scan_kernel<<<256, 256, 0, stream>>>(dtf, xdbl, xi, xz, A_log, Dp, yg_bf);
    // 6) out_proj: [4096,2048] x [1024,2048]^T -> mbuf[4096,1024]
    gemm_bf16_wmma<false, false><<<dim3(16, 64), blk, 0, stream>>>(
        yg_bf, 2048, wout_bf, 2048, nullptr, mbuf, nullptr, 1024, 1024, 2048);
    // 7) residual + LayerNorm
    ln_residual_kernel<<<4096, 256, 0, stream>>>(x, mbuf, lnw, lnb, out);
}
